// BahdanauAttention_41790031790660
// MI455X (gfx1250) — matrix operand order repaired and checked by reading, not compiled, not run on hardware
//
#include <hip/hip_runtime.h>
#include <hip/hip_bf16.h>

// ---------------------------------------------------------------------------
// Bahdanau attention, B=32 S=2048 H=E=1024, fp32 in/out.
// Dominant cost: enc@W_e (137 GFLOP) fused with tanh + v-dot so the 256MB
// "energy" tensor is never materialized. bf16 WMMA (v_wmma_f32_16x16x32_bf16),
// fp32 accumulate. W_e is transposed+converted to bf16 once (2MB, L2-resident).
// M-tile = 32 tokens per WG -> 2 WMMAs per B-fragment load (halves L2 traffic
// per FLOP vs M=16). Second pass (context) is a pure 256MB bandwidth stream.
// ---------------------------------------------------------------------------

#define Bn 32
#define Sn 2048
#define Hn 1024
#define NEGV -10000000000.0f

typedef __attribute__((ext_vector_type(16))) __bf16 v16bf;
typedef __attribute__((ext_vector_type(8)))  float  v8f;

union BF16Frag {
    v16bf v;
    uint4 q[2];
};

__device__ __forceinline__ unsigned int f32_to_bf16_bits(float f) {
    unsigned int u = __float_as_uint(f);
    u += 0x7FFFu + ((u >> 16) & 1u);   // round to nearest even
    return u >> 16;
}

// ---------------------------------------------------------------------------
// Prep 1: W_eT[n][k] = bf16(W_e[k][n])   (1024x1024, one-time 4MB read)
// ---------------------------------------------------------------------------
__global__ __launch_bounds__(256) void wet_kernel(const float* __restrict__ W_e,
                                                  unsigned short* __restrict__ weT) {
    int o = blockIdx.x * 256 + threadIdx.x;   // 0 .. 1M-1, coalesced writes
    int n = o >> 10;
    int k = o & 1023;
    weT[o] = (unsigned short)f32_to_bf16_bits(W_e[k * Hn + n]);
}

// ---------------------------------------------------------------------------
// Prep 2: hproj[b][j] = hidden[b]@W_h[:,j] + b_h[j] + b_e[j]   (tiny GEMM)
// ---------------------------------------------------------------------------
__global__ __launch_bounds__(256) void hproj_kernel(const float* __restrict__ hidden,
                                                    const float* __restrict__ W_h,
                                                    const float* __restrict__ b_h,
                                                    const float* __restrict__ b_e,
                                                    float* __restrict__ hproj) {
    int idx = blockIdx.x * 256 + threadIdx.x;  // 0 .. 32*1024-1
    int b = idx >> 10;
    int j = idx & 1023;
    const float* hrow = hidden + b * Hn;
    float acc = 0.f;
    for (int k = 0; k < Hn; ++k)
        acc = fmaf(hrow[k], W_h[k * Hn + j], acc);   // W_h reads coalesced in j
    hproj[idx] = acc + b_h[j] + b_e[j];
}

// ---------------------------------------------------------------------------
// Main pass:
//   attn_raw[b][s] = sum_n tanh( (enc[b,s,:]@W_e)[n] + hproj[b][n] ) * v[n]
// WG = 256 threads = 8 waves; WG tile = 32 tokens x 1024 N; wave = 32 x 128.
// K loop step 32. A (fp32->bf16) staged via LDS; B fragments from bf16 W_eT.
// Two 16-row A fragments per lane -> 2 WMMAs per B-fragment load.
// ---------------------------------------------------------------------------
__global__ __launch_bounds__(256) void energy_kernel(const float* __restrict__ enc,
                                                     const unsigned short* __restrict__ weT,
                                                     const float* __restrict__ hproj,
                                                     const float* __restrict__ vvec,
                                                     float* __restrict__ attn_raw) {
    const int b    = blockIdx.x >> 6;          // 32 batches
    const int s0   = (blockIdx.x & 63) << 5;   // 64 token tiles of 32
    const int tid  = threadIdx.x;
    const int lane = tid & 31;
    const int wave = tid >> 5;
    const int col  = lane & 15;   // N within tile / M row of A frag
    const int half = lane >> 4;   // K-half selector

    __shared__ alignas(16) unsigned short Atile[32][40];  // 32x32 bf16, padded stride
    __shared__ float attn_s[32];

    v8f acc[16] = {};   // [nt*2 + mtile]: 8 N-tiles x 2 M-tiles of 16x16 f32

    // Cooperative A staging: thread t -> row t>>3 (0..31), cols 4*(t&7)..+3
    const int arow = tid >> 3;
    const int acol = (tid & 7) * 4;
    const float* aptr = enc + ((size_t)(b * Sn + s0 + arow)) * Hn + acol;

    for (int k0 = 0; k0 < Hn; k0 += 32) {
        float4 f4 = *(const float4*)(aptr + k0);
        unsigned int p0 = f32_to_bf16_bits(f4.x) | (f32_to_bf16_bits(f4.y) << 16);
        unsigned int p1 = f32_to_bf16_bits(f4.z) | (f32_to_bf16_bits(f4.w) << 16);
        *reinterpret_cast<uint2*>(&Atile[arow][acol]) = make_uint2(p0, p1);
        __syncthreads();

        // A fragments (16-bit 16x32 layout): lane -> M = col (+16 for tile 1);
        // e0..7 : K = half*8 + 0..7    (contiguous 16B)
        // e8..15: K = 16 + half*8 + 0..7 (contiguous 16B)
        BF16Frag af0, af1;
        af0.q[0] = *reinterpret_cast<const uint4*>(&Atile[col][half * 8]);
        af0.q[1] = *reinterpret_cast<const uint4*>(&Atile[col][16 + half * 8]);
        af1.q[0] = *reinterpret_cast<const uint4*>(&Atile[col + 16][half * 8]);
        af1.q[1] = *reinterpret_cast<const uint4*>(&Atile[col + 16][16 + half * 8]);

#pragma unroll
        for (int nt = 0; nt < 8; ++nt) {
            // B fragment (32x16): lane -> N = col, K = k0 + half*8 + 0..7, k0 + 16 + half*8 + 0..7
            const int n = wave * 128 + nt * 16 + col;
            BF16Frag bf;
            const uint4* pb =
                reinterpret_cast<const uint4*>(weT + (size_t)n * Hn + k0 + half * 8);
            bf.q[0] = pb[0];
            bf.q[1] = pb[2];
            acc[nt * 2 + 0] = __builtin_amdgcn_wmma_f32_16x16x32_bf16(
                false, af0.v, false, bf.v, (short)0, acc[nt * 2 + 0], false, false);
            acc[nt * 2 + 1] = __builtin_amdgcn_wmma_f32_16x16x32_bf16(
                false, af1.v, false, bf.v, (short)0, acc[nt * 2 + 1], false, false);
        }
        __syncthreads();
    }

    // Epilogue: tanh + dot with v.
    // D layout: VGPR r -> M = mtile*16 + half*8 + r, N = col.
    float p[2][8];
#pragma unroll
    for (int t = 0; t < 2; ++t)
#pragma unroll
        for (int r = 0; r < 8; ++r) p[t][r] = 0.f;

#pragma unroll
    for (int nt = 0; nt < 8; ++nt) {
        const int n = wave * 128 + nt * 16 + col;
        const float hp = hproj[b * Hn + n];
        const float vv = vvec[n];
#pragma unroll
        for (int t = 0; t < 2; ++t) {
#pragma unroll
            for (int r = 0; r < 8; ++r) {
                float e = tanhf(acc[nt * 2 + t][r] + hp);
                p[t][r] = fmaf(e, vv, p[t][r]);
            }
        }
    }
    // Reduce over the 16 N-lanes within each half (halves hold different M rows)
#pragma unroll
    for (int t = 0; t < 2; ++t) {
#pragma unroll
        for (int r = 0; r < 8; ++r) {
            float x = p[t][r];
            x += __shfl_xor(x, 1);
            x += __shfl_xor(x, 2);
            x += __shfl_xor(x, 4);
            x += __shfl_xor(x, 8);
            p[t][r] = x;
        }
    }
    if (tid < 32) attn_s[tid] = 0.f;
    __syncthreads();
    if (col == 0) {
#pragma unroll
        for (int t = 0; t < 2; ++t)
#pragma unroll
            for (int r = 0; r < 8; ++r)
                atomicAdd(&attn_s[t * 16 + half * 8 + r], p[t][r]);
    }
    __syncthreads();
    if (tid < 32) attn_raw[b * Sn + s0 + tid] = attn_s[tid];
}

// ---------------------------------------------------------------------------
// Masked softmax over S per batch; also zeroes the context region of d_out.
// out layout: [0, 32768) = context, [32768, 98304) = attention_weights.
// ---------------------------------------------------------------------------
__global__ __launch_bounds__(256) void softmax_kernel(const float* __restrict__ attn_raw,
                                                      const int* __restrict__ mask,
                                                      float* __restrict__ out) {
    const int b = blockIdx.x;
    const int tid = threadIdx.x;
    __shared__ float red[256];

    float vals[8];
    float mx = -3.4e38f;
#pragma unroll
    for (int i = 0; i < 8; ++i) {
        int s = tid + i * 256;
        float a = attn_raw[b * Sn + s];
        if (mask[b * Sn + s] == 0) a = NEGV;
        vals[i] = a;
        mx = fmaxf(mx, a);
    }
    red[tid] = mx;
    __syncthreads();
    for (int off = 128; off > 0; off >>= 1) {
        if (tid < off) red[tid] = fmaxf(red[tid], red[tid + off]);
        __syncthreads();
    }
    mx = red[0];
    __syncthreads();

    float sum = 0.f;
#pragma unroll
    for (int i = 0; i < 8; ++i) {
        vals[i] = __expf(vals[i] - mx);
        sum += vals[i];
    }
    red[tid] = sum;
    __syncthreads();
    for (int off = 128; off > 0; off >>= 1) {
        if (tid < off) red[tid] += red[tid + off];
        __syncthreads();
    }
    const float inv = 1.0f / red[0];

#pragma unroll
    for (int i = 0; i < 8; ++i)
        out[Bn * Hn + b * Sn + tid + i * 256] = vals[i] * inv;

    // zero the context region (harness does not re-poison between replays)
    for (int j = tid; j < Hn; j += 256) out[b * Hn + j] = 0.f;
}

// ---------------------------------------------------------------------------
// context[b][e] = sum_s w[b][s] * enc[b][s][e]  — pure 256MB stream.
// 512 WGs: (b, 128-row s-chunk); thread owns 4 e-columns; global f32 atomics.
// ---------------------------------------------------------------------------
__global__ __launch_bounds__(256) void context_kernel(const float* __restrict__ enc,
                                                      const float* __restrict__ w,
                                                      float* __restrict__ ctx) {
    const int blk = blockIdx.x;
    const int b = blk >> 4;
    const int sc = blk & 15;
    const int tid = threadIdx.x;

    const float* wrow = w + b * Sn + sc * 128;
    const float* base = enc + ((size_t)(b * Sn + sc * 128)) * Hn + tid * 4;

    float4 acc = make_float4(0.f, 0.f, 0.f, 0.f);
    for (int s = 0; s < 128; ++s) {
        float ws = wrow[s];
        float4 f = *(const float4*)(base + (size_t)s * Hn);
        acc.x = fmaf(ws, f.x, acc.x);
        acc.y = fmaf(ws, f.y, acc.y);
        acc.z = fmaf(ws, f.z, acc.z);
        acc.w = fmaf(ws, f.w, acc.w);
    }
    float* c = ctx + b * Hn + tid * 4;
    atomicAdd(c + 0, acc.x);
    atomicAdd(c + 1, acc.y);
    atomicAdd(c + 2, acc.z);
    atomicAdd(c + 3, acc.w);
}

extern "C" void kernel_launch(void* const* d_in, const int* in_sizes, int n_in,
                              void* d_out, int out_size, void* d_ws, size_t ws_size,
                              hipStream_t stream) {
    const float* hidden = (const float*)d_in[0];
    const float* enc    = (const float*)d_in[1];
    const int*   mask   = (const int*)d_in[2];
    const float* W_h    = (const float*)d_in[3];
    const float* b_h    = (const float*)d_in[4];
    const float* W_e    = (const float*)d_in[5];
    const float* b_e    = (const float*)d_in[6];
    const float* v      = (const float*)d_in[7];
    float* out = (float*)d_out;

    char* ws = (char*)d_ws;
    unsigned short* weT  = (unsigned short*)ws;                         // 2 MB bf16 W_e^T
    float* hproj         = (float*)(ws + (size_t)2 * 1024 * 1024);      // 128 KB
    float* attn_raw      = (float*)(ws + (size_t)2 * 1024 * 1024
                                       + (size_t)Bn * Hn * 4);          // 256 KB

    wet_kernel<<<(Hn * Hn) / 256, 256, 0, stream>>>(W_e, weT);
    hproj_kernel<<<(Bn * Hn) / 256, 256, 0, stream>>>(hidden, W_h, b_h, b_e, hproj);
    energy_kernel<<<Bn * (Sn / 32), 256, 0, stream>>>(enc, weT, hproj, v, attn_raw);
    softmax_kernel<<<Bn, 256, 0, stream>>>(attn_raw, mask, out);
    context_kernel<<<Bn * (Sn / 128), 256, 0, stream>>>(enc, out + Bn * Hn, out);
}
